// LCMR3S_73684458930894
// MI455X (gfx1250) — matrix operand order repaired and checked by reading, not compiled, not run on hardware
//
#include <hip/hip_runtime.h>
#include <hip/hip_bf16.h>

// ---- CDNA5 WMMA vector types -------------------------------------------------
typedef __attribute__((ext_vector_type(16))) __bf16 v16bf;
typedef __attribute__((ext_vector_type(8)))  float  v8f;
typedef __attribute__((ext_vector_type(2)))  float  v2f;

// ---- bf16 helpers (RNE) ------------------------------------------------------
__device__ __forceinline__ unsigned short f2bfu(float f) {
    unsigned u = __builtin_bit_cast(unsigned, f);
    unsigned r = u + 0x7fffu + ((u >> 16) & 1u);
    return (unsigned short)(r >> 16);
}
__device__ __forceinline__ __bf16 u2bf(unsigned short u) {
    return __builtin_bit_cast(__bf16, u);
}

// =============================================================================
// f32 -> bf16 convert
// =============================================================================
__global__ void k_f2bf(const float* __restrict__ s, unsigned short* __restrict__ d, int n) {
    int i = blockIdx.x * blockDim.x + threadIdx.x;
    if (i < n) d[i] = f2bfu(s[i]);
}

// =============================================================================
// LayerNorm over 256 cols, one wave per row, output bf16
// =============================================================================
__global__ void __launch_bounds__(256) k_ln(const float* __restrict__ x,
                                            const float* __restrict__ g,
                                            const float* __restrict__ be,
                                            unsigned short* __restrict__ out) {
    const int lane = threadIdx.x & 31;
    const int row  = blockIdx.x * (blockDim.x >> 5) + (threadIdx.x >> 5);
    const float* xr = x + (size_t)row * 256;
    float v[8]; float s = 0.f, s2 = 0.f;
#pragma unroll
    for (int j = 0; j < 8; ++j) { v[j] = xr[lane + 32 * j]; s += v[j]; s2 += v[j] * v[j]; }
#pragma unroll
    for (int m = 1; m < 32; m <<= 1) { s += __shfl_xor(s, m, 32); s2 += __shfl_xor(s2, m, 32); }
    float mean = s * (1.0f / 256.0f);
    float var  = s2 * (1.0f / 256.0f) - mean * mean;
    float rstd = rsqrtf(var + 1e-5f);
#pragma unroll
    for (int j = 0; j < 8; ++j) {
        int c = lane + 32 * j;
        out[(size_t)row * 256 + c] = f2bfu((v[j] - mean) * rstd * g[c] + be[c]);
    }
}

// =============================================================================
// Generic bf16 GEMM: C[M,N] = A[M,K] @ W[N,K]^T, wave tile = 32(M) x 64(N):
// 2 A-fragments, 8 accumulators, each B-fragment feeds 2 WMMAs (halves B traffic).
// epi: 0=textproj(+bias+pos->f32), 1=qkv scatter(bf16 q/k/vT), 2,4=residual add f32,
//      3=gelu->bf16
// =============================================================================
__global__ void __launch_bounds__(256) k_gemm(const unsigned short* __restrict__ A,
                                              const unsigned short* __restrict__ Bw,
                                              const float* __restrict__ bias,
                                              int M, int N, int K, int epi,
                                              float* __restrict__ outF,
                                              unsigned short* __restrict__ outBF,
                                              const float* __restrict__ pos,
                                              unsigned short* __restrict__ qo,
                                              unsigned short* __restrict__ ko,
                                              unsigned short* __restrict__ vTo) {
    const int lane   = threadIdx.x & 31;
    const int wave   = blockIdx.x * (blockDim.x >> 5) + (threadIdx.x >> 5);
    const int ntiles = N >> 6;
    const int mt = wave / ntiles;
    const int nt = wave % ntiles;
    const int m0 = mt << 5;             // 32-row tile
    const int n0 = nt << 6;
    const int half = lane >> 4;
    const int lr   = lane & 15;

    v8f z = {};
    v8f accA[4]; accA[0] = z; accA[1] = z; accA[2] = z; accA[3] = z;
    v8f accB[4]; accB[0] = z; accB[1] = z; accB[2] = z; accB[3] = z;

    const unsigned short* ArowA = A + (size_t)(m0 + lr) * K;
    const unsigned short* ArowB = A + (size_t)(m0 + 16 + lr) * K;

    for (int kc = 0; kc < K; kc += 32) {
        // prefetch the streamed A operand one chunk ahead (global_prefetch_b8)
        __builtin_prefetch(ArowA + kc + 32, 0, 3);
        __builtin_prefetch(ArowB + kc + 32, 0, 3);
        // A fragments (16x32 bf16, contiguous k = 16*half + e, same as B)
        v16bf afA, afB;
#pragma unroll
        for (int e = 0; e < 16; ++e) {
            int kk = kc + half * 16 + e;
            afA[e] = u2bf(ArowA[kk]);
            afB[e] = u2bf(ArowB[kk]);
        }
#pragma unroll
        for (int s = 0; s < 4; ++s) {
            int n = n0 + s * 16 + lr;
            const unsigned short* Brow = Bw + (size_t)n * K;
            v16bf bf;
#pragma unroll
            for (int e = 0; e < 16; ++e) bf[e] = u2bf(Brow[kc + half * 16 + e]);
            accA[s] = __builtin_amdgcn_wmma_f32_16x16x32_bf16(false, afA, false, bf,
                                                              (short)0, accA[s], false, false);
            accB[s] = __builtin_amdgcn_wmma_f32_16x16x32_bf16(false, afB, false, bf,
                                                              (short)0, accB[s], false, false);
        }
    }

#pragma unroll
    for (int g = 0; g < 2; ++g) {
#pragma unroll
        for (int s = 0; s < 4; ++s) {
            int n = n0 + s * 16 + lr;
            float bv = bias ? bias[n] : 0.0f;
            v8f av = g ? accB[s] : accA[s];
#pragma unroll
            for (int r = 0; r < 8; ++r) {
                int row = m0 + g * 16 + r + half * 8;
                float v = av[r] + bv;
                if (epi == 0) {
                    int si = row & 1023;
                    outF[(size_t)row * 256 + n] = v + pos[si * 256 + n];
                } else if (epi == 1) {
                    int b = row >> 10, si = row & 1023;
                    if (n < 256) {
                        int h = n >> 5, d = n & 31;
                        qo[(((size_t)(b * 8 + h) * 1024 + si) << 5) + d] = f2bfu(v);
                    } else if (n < 512) {
                        int n2 = n - 256; int h = n2 >> 5, d = n2 & 31;
                        ko[(((size_t)(b * 8 + h) * 1024 + si) << 5) + d] = f2bfu(v);
                    } else {
                        int n2 = n - 512; int h = n2 >> 5, d = n2 & 31;
                        vTo[(((size_t)(b * 8 + h) * 32 + d) << 10) + si] = f2bfu(v);
                    }
                } else if (epi == 2 || epi == 4) {
                    outF[(size_t)row * 256 + n] += v;
                } else { // epi == 3 : exact GELU -> bf16
                    float gl = 0.5f * v * (1.0f + erff(v * 0.70710678118654752f));
                    outBF[(size_t)row * (size_t)N + n] = f2bfu(gl);
                }
            }
        }
    }
}

// =============================================================================
// Flash attention: wave = one (b,h,16-query tile); keys in chunks of 32.
// q,k: [B,H,S,32] bf16 ; vT: [B,H,32,S] bf16 ; ctx: [B,S,256] bf16
// =============================================================================
__global__ void __launch_bounds__(256) k_attn(const unsigned short* __restrict__ q,
                                              const unsigned short* __restrict__ kk,
                                              const unsigned short* __restrict__ vT,
                                              unsigned short* __restrict__ ctx) {
    __shared__ unsigned short Pt[8][16][32];
    const int lane = threadIdx.x & 31;
    const int wv   = threadIdx.x >> 5;
    const int gw   = blockIdx.x * 8 + wv;
    const int qt   = gw & 63;
    const int h    = (gw >> 6) & 7;
    const int b    = gw >> 9;
    const int half = lane >> 4;
    const int lr   = lane & 15;
    const size_t bh = (size_t)(b * 8 + h);
    const unsigned short* qp = q  + (bh * 1024 + (size_t)qt * 16) * 32;
    const unsigned short* kp = kk + bh * 1024 * 32;
    const unsigned short* vp = vT + bh * 32 * 1024;

    v16bf aq;
#pragma unroll
    for (int e = 0; e < 16; ++e) {
        int d = half * 16 + e;
        aq[e] = u2bf(qp[lr * 32 + d]);
    }
    float mi[8], li[8];
    v8f acc0 = {}, acc1 = {};
#pragma unroll
    for (int r = 0; r < 8; ++r) { mi[r] = -1e30f; li[r] = 0.0f; }
    const float rs = 0.17677669529663689f; // 1/sqrt(32)

    for (int kc = 0; kc < 1024; kc += 32) {
        v8f sc0 = {}, sc1 = {};
        {
            v16bf bf;
#pragma unroll
            for (int e = 0; e < 16; ++e) bf[e] = u2bf(kp[(kc + lr) * 32 + half * 16 + e]);
            sc0 = __builtin_amdgcn_wmma_f32_16x16x32_bf16(false, aq, false, bf, (short)0, sc0, false, false);
#pragma unroll
            for (int e = 0; e < 16; ++e) bf[e] = u2bf(kp[(kc + 16 + lr) * 32 + half * 16 + e]);
            sc1 = __builtin_amdgcn_wmma_f32_16x16x32_bf16(false, aq, false, bf, (short)0, sc1, false, false);
        }
        // online softmax per row (rows live across the 16 lanes of each half-wave)
#pragma unroll
        for (int r = 0; r < 8; ++r) {
            float s0 = sc0[r] * rs, s1 = sc1[r] * rs;
            float cm = fmaxf(s0, s1);
#pragma unroll
            for (int msk = 1; msk < 16; msk <<= 1) cm = fmaxf(cm, __shfl_xor(cm, msk, 32));
            float mn = fmaxf(mi[r], cm);
            float alpha = __expf(mi[r] - mn);
            float p0 = __expf(s0 - mn), p1 = __expf(s1 - mn);
            float rsum = p0 + p1;
#pragma unroll
            for (int msk = 1; msk < 16; msk <<= 1) rsum += __shfl_xor(rsum, msk, 32);
            li[r] = li[r] * alpha + rsum;
            mi[r] = mn;
            acc0[r] = acc0[r] * alpha;
            acc1[r] = acc1[r] * alpha;
            Pt[wv][r + half * 8][lr]      = f2bfu(p0);
            Pt[wv][r + half * 8][lr + 16] = f2bfu(p1);
        }
        asm volatile("s_wait_dscnt 0" ::: "memory");
        v16bf ap;
#pragma unroll
        for (int e = 0; e < 16; ++e) {
            int kof = half * 16 + e;
            ap[e] = u2bf(Pt[wv][lr][kof]);
        }
        {
            v16bf bf;
#pragma unroll
            for (int e = 0; e < 16; ++e) bf[e] = u2bf(vp[(size_t)lr * 1024 + kc + half * 16 + e]);
            acc0 = __builtin_amdgcn_wmma_f32_16x16x32_bf16(false, ap, false, bf, (short)0, acc0, false, false);
#pragma unroll
            for (int e = 0; e < 16; ++e) bf[e] = u2bf(vp[(size_t)(lr + 16) * 1024 + kc + half * 16 + e]);
            acc1 = __builtin_amdgcn_wmma_f32_16x16x32_bf16(false, ap, false, bf, (short)0, acc1, false, false);
        }
    }
#pragma unroll
    for (int r = 0; r < 8; ++r) {
        float inv = 1.0f / li[r];
        int srow = qt * 16 + r + half * 8;
        size_t base = ((size_t)b * 1024 + srow) * 256 + h * 32;
        ctx[base + lr]      = f2bfu(acc0[r] * inv);
        ctx[base + 16 + lr] = f2bfu(acc1[r] * inv);
    }
}

// =============================================================================
// RNN scan (1024 sequential steps) + ssm + logits. One workgroup, 16 waves.
// Full fp32 precision via v_wmma_f32_16x16x4_f32 (M=16 batch, N=16/wave, K=512).
// x_{t+1} is prefetched (global_prefetch_b8) before the step-t WMMA loop.
// =============================================================================
__global__ void __launch_bounds__(512) k_rnn(const float* __restrict__ x,
                                             const float* __restrict__ Wi,
                                             const float* __restrict__ bi,
                                             const float* __restrict__ Wo,
                                             const float* __restrict__ bo,
                                             const float* __restrict__ cw,
                                             const float* __restrict__ cb,
                                             const float* __restrict__ h0,
                                             float* __restrict__ ssm_out,
                                             float* __restrict__ out01) {
    __shared__ float xh[16][512];   // [batch][x_t (0:256) | h (256:512)]
    __shared__ float sm[16][256];
    const int tid  = threadIdx.x;
    const int lane = tid & 31, wv = tid >> 5;
    const int half = lane >> 4, lr = lane & 15;
    const int n    = wv * 16 + lr;

    for (int i = tid; i < 16 * 256; i += 512) xh[i >> 8][256 + (i & 255)] = h0[i];
    for (int i = tid; i < 16 * 256; i += 512) {
        int bb = i >> 8, e = i & 255;
        xh[bb][e] = x[((size_t)bb * 1024) * 256 + e];
    }
    __syncthreads();

    const float bias = bi[n];
    const float* wrow = Wi + (size_t)n * 512;
    const int koff = half * 2;

    for (int t = 0; t < 1024; ++t) {
        // prefetch next timestep's 16KB x-slab (128 cachelines) to overlap with WMMA
        if (t < 1023 && tid < 128) {
            int bb = tid >> 3, e = (tid & 7) * 32;
            __builtin_prefetch(&x[((size_t)bb * 1024 + (t + 1)) * 256 + e], 0, 3);
        }
        v8f a0 = {}, a1 = {}, a2 = {}, a3 = {};
        for (int kc = 0; kc < 512; kc += 16) {
            v2f av, bv;
            int kb = kc + koff;
            av[0] = xh[lr][kb];     av[1] = xh[lr][kb + 1];
            bv[0] = wrow[kb];       bv[1] = wrow[kb + 1];
            a0 = __builtin_amdgcn_wmma_f32_16x16x4_f32(false, av, false, bv, (short)0, a0, false, false);
            kb += 4;
            av[0] = xh[lr][kb];     av[1] = xh[lr][kb + 1];
            bv[0] = wrow[kb];       bv[1] = wrow[kb + 1];
            a1 = __builtin_amdgcn_wmma_f32_16x16x4_f32(false, av, false, bv, (short)0, a1, false, false);
            kb += 4;
            av[0] = xh[lr][kb];     av[1] = xh[lr][kb + 1];
            bv[0] = wrow[kb];       bv[1] = wrow[kb + 1];
            a2 = __builtin_amdgcn_wmma_f32_16x16x4_f32(false, av, false, bv, (short)0, a2, false, false);
            kb += 4;
            av[0] = xh[lr][kb];     av[1] = xh[lr][kb + 1];
            bv[0] = wrow[kb];       bv[1] = wrow[kb + 1];
            a3 = __builtin_amdgcn_wmma_f32_16x16x4_f32(false, av, false, bv, (short)0, a3, false, false);
        }
        __syncthreads();   // all waves finished reading xh
#pragma unroll
        for (int r = 0; r < 8; ++r) {
            int bb = r + half * 8;
            float v = a0[r] + a1[r] + a2[r] + a3[r] + bias;
            float th = tanhf(v);
            xh[bb][256 + n] = (th > 0.0f) ? th : 0.0f;
        }
        if (t < 1023) {
            for (int i = tid; i < 16 * 256; i += 512) {
                int bb = i >> 8, e = i & 255;
                xh[bb][e] = x[((size_t)bb * 1024 + (t + 1)) * 256 + e];
            }
        }
        __syncthreads();
    }

    // ssm = h @ Wo^T + bo
    {
        v8f a0 = {}, a1 = {}, a2 = {}, a3 = {};
        const float* orow = Wo + (size_t)n * 256;
        for (int kc = 0; kc < 256; kc += 16) {
            v2f av, bv;
            int kb = kc + koff;
            av[0] = xh[lr][256 + kb]; av[1] = xh[lr][256 + kb + 1];
            bv[0] = orow[kb];         bv[1] = orow[kb + 1];
            a0 = __builtin_amdgcn_wmma_f32_16x16x4_f32(false, av, false, bv, (short)0, a0, false, false);
            kb += 4;
            av[0] = xh[lr][256 + kb]; av[1] = xh[lr][256 + kb + 1];
            bv[0] = orow[kb];         bv[1] = orow[kb + 1];
            a1 = __builtin_amdgcn_wmma_f32_16x16x4_f32(false, av, false, bv, (short)0, a1, false, false);
            kb += 4;
            av[0] = xh[lr][256 + kb]; av[1] = xh[lr][256 + kb + 1];
            bv[0] = orow[kb];         bv[1] = orow[kb + 1];
            a2 = __builtin_amdgcn_wmma_f32_16x16x4_f32(false, av, false, bv, (short)0, a2, false, false);
            kb += 4;
            av[0] = xh[lr][256 + kb]; av[1] = xh[lr][256 + kb + 1];
            bv[0] = orow[kb];         bv[1] = orow[kb + 1];
            a3 = __builtin_amdgcn_wmma_f32_16x16x4_f32(false, av, false, bv, (short)0, a3, false, false);
        }
#pragma unroll
        for (int r = 0; r < 8; ++r) {
            int bb = r + half * 8;
            float v = a0[r] + a1[r] + a2[r] + a3[r] + bo[n];
            sm[bb][n] = v;
            ssm_out[(size_t)bb * 256 + n] = v;
        }
    }
    __syncthreads();
    if (tid < 16) {
        float d = cb[0];
        for (int e = 0; e < 256; ++e) d += sm[tid][e] * cw[e];
        out01[tid]      = d;                              // logits
        out01[16 + tid] = 1.0f / (1.0f + __expf(-d));     // probas
    }
}

// =============================================================================
// Host launcher
// =============================================================================
extern "C" void kernel_launch(void* const* d_in, const int* in_sizes, int n_in,
                              void* d_out, int out_size, void* d_ws, size_t ws_size,
                              hipStream_t stream) {
    (void)in_sizes; (void)n_in; (void)out_size; (void)ws_size;
    const float* text_emb = (const float*)d_in[0];
    const float* pos_emb  = (const float*)d_in[4];
    const float* text_W   = (const float*)d_in[5];
    const float* text_b   = (const float*)d_in[6];
    const float* ln1_g    = (const float*)d_in[9];
    const float* ln1_b    = (const float*)d_in[10];
    const float* qkv_W    = (const float*)d_in[11];
    const float* qkv_b    = (const float*)d_in[12];
    const float* out_W    = (const float*)d_in[13];
    const float* out_b    = (const float*)d_in[14];
    const float* ln2_g    = (const float*)d_in[15];
    const float* ln2_b    = (const float*)d_in[16];
    const float* ffn_W1   = (const float*)d_in[17];
    const float* ffn_b1   = (const float*)d_in[18];
    const float* ffn_W2   = (const float*)d_in[19];
    const float* ffn_b2   = (const float*)d_in[20];
    const float* cls_W    = (const float*)d_in[21];
    const float* cls_b    = (const float*)d_in[22];
    const float* i2h_W    = (const float*)d_in[23];
    const float* i2h_b    = (const float*)d_in[24];
    const float* h2o_W    = (const float*)d_in[25];
    const float* h2o_b    = (const float*)d_in[26];
    const float* h0       = (const float*)d_in[27];

    float* out = (float*)d_out;
    float* x   = out + 32;                       // final_vector [16,1024,256]
    float* ssm = out + 32 + 16 * 1024 * 256;     // [16,256]

    // workspace layout (bf16 buffers)
    unsigned short* te_bf  = (unsigned short*)d_ws;
    unsigned short* xn_bf  = te_bf  + (size_t)16384 * 768;
    unsigned short* q_bf   = xn_bf  + (size_t)16384 * 256;
    unsigned short* k_bf   = q_bf   + (size_t)4194304;
    unsigned short* vT_bf  = k_bf   + (size_t)4194304;
    unsigned short* ctx_bf = vT_bf  + (size_t)4194304;
    unsigned short* h1_bf  = ctx_bf + (size_t)16384 * 256;
    unsigned short* tw_bf  = h1_bf  + (size_t)16384 * 1024;
    unsigned short* qw_bf  = tw_bf  + (size_t)196608;
    unsigned short* ow_bf  = qw_bf  + (size_t)393216;
    unsigned short* f1_bf  = ow_bf  + (size_t)131072;
    unsigned short* f2_bf  = f1_bf  + (size_t)524288;

    auto cvt = [&](const float* s, unsigned short* d, int n) {
        k_f2bf<<<(n + 255) / 256, 256, 0, stream>>>(s, d, n);
    };
    cvt(text_emb, te_bf, 16384 * 768);
    cvt(text_W,   tw_bf, 256 * 768);
    cvt(qkv_W,    qw_bf, 2 * 768 * 256);
    cvt(out_W,    ow_bf, 2 * 256 * 256);
    cvt(ffn_W1,   f1_bf, 2 * 1024 * 256);
    cvt(ffn_W2,   f2_bf, 2 * 256 * 1024);

    // waves = (M/32)*(N/64), 8 waves per 256-thread block
    // lang = text_emb @ text_W^T + text_b + pos  -> x (f32 in d_out)
    k_gemm<<<256, 256, 0, stream>>>(te_bf, tw_bf, text_b, 16384, 256, 768, 0,
                                    x, nullptr, pos_emb, nullptr, nullptr, nullptr);

    for (int l = 0; l < 2; ++l) {
        k_ln<<<2048, 256, 0, stream>>>(x, ln1_g + l * 256, ln1_b + l * 256, xn_bf);
        k_gemm<<<768, 256, 0, stream>>>(xn_bf, qw_bf + (size_t)l * 768 * 256, qkv_b + l * 768,
                                        16384, 768, 256, 1,
                                        nullptr, nullptr, nullptr, q_bf, k_bf, vT_bf);
        k_attn<<<1024, 256, 0, stream>>>(q_bf, k_bf, vT_bf, ctx_bf);
        k_gemm<<<256, 256, 0, stream>>>(ctx_bf, ow_bf + (size_t)l * 256 * 256, out_b + l * 256,
                                        16384, 256, 256, 2,
                                        x, nullptr, nullptr, nullptr, nullptr, nullptr);
        k_ln<<<2048, 256, 0, stream>>>(x, ln2_g + l * 256, ln2_b + l * 256, xn_bf);
        k_gemm<<<1024, 256, 0, stream>>>(xn_bf, f1_bf + (size_t)l * 1024 * 256, ffn_b1 + l * 1024,
                                         16384, 1024, 256, 3,
                                         nullptr, h1_bf, nullptr, nullptr, nullptr, nullptr);
        k_gemm<<<256, 256, 0, stream>>>(h1_bf, f2_bf + (size_t)l * 256 * 1024, ffn_b2 + l * 256,
                                        16384, 256, 1024, 4,
                                        x, nullptr, nullptr, nullptr, nullptr, nullptr);
    }

    k_rnn<<<1, 512, 0, stream>>>(x, i2h_W, i2h_b, h2o_W, h2o_b, cls_W, cls_b, h0, ssm, out);
}
